// OptimalTransportLoss_80539226734653
// MI455X (gfx1250) — matrix operand order repaired and checked by reading, not compiled, not run on hardware
//
#include <hip/hip_runtime.h>

// Optimal-transport loss (Sinkhorn + Gromov-Wasserstein) for MI455X / gfx1250.
// GEMMs: v_wmma_f32_16x16x32_bf16 (wave32). Big 4096^3 GEMMs use the Tensor
// Data Mover (tensor_load_to_lds) with double-buffered LDS staging.

constexpr int   NN      = 4096;
constexpr int   DD      = 256;
constexpr float REG_INV = 100.0f;        // 1/REG, REG = 0.01
constexpr float EPS_S   = 1e-8f;
constexpr int   ITERS   = 50;
constexpr float MU      = 1.0f / 4096.0f;

typedef __bf16 bf16_t;
typedef __attribute__((ext_vector_type(16))) __bf16   v16bf;
typedef __attribute__((ext_vector_type(8)))  float    v8f;
typedef __attribute__((ext_vector_type(8)))  __bf16   bf16x8;
typedef __attribute__((ext_vector_type(4)))  unsigned u32x4;
typedef __attribute__((ext_vector_type(8)))  int      i32x8;
typedef __attribute__((ext_vector_type(4)))  int      i32x4;

// ---------------- WMMA fragment loaders (wave32, 16x16x32 bf16) ----------------
// A (16x32 MxK): lane L<16 -> row M=L, K={0..7,16..23}; lane>=16 -> M=L-16, K={8..15,24..31}
__device__ __forceinline__ v16bf load_a_frag(const bf16_t* A, int lda, int i0, int k0, int lane) {
  const bf16_t* p = A + (size_t)(i0 + (lane & 15)) * lda + (size_t)(k0 + ((lane >> 4) << 3));
  bf16x8 lo = *(const bf16x8*)p;
  bf16x8 hi = *(const bf16x8*)(p + 16);
  v16bf r;
#pragma unroll
  for (int t = 0; t < 8; ++t) { r[t] = lo[t]; r[t + 8] = hi[t]; }
  return r;
}
// B (32x16 KxN) with B[k,n] = Y[j0+n, k] (Y row-major): lane n reads K={0..7,16..23} (lane>=16: {8..15,24..31}).
__device__ __forceinline__ v16bf load_bT_frag(const bf16_t* Y, int ldy, int j0, int k0, int lane) {
  const bf16_t* p = Y + (size_t)(j0 + (lane & 15)) * ldy + (size_t)(k0 + ((lane >> 4) << 3));
  bf16x8 lo = *(const bf16x8*)p;
  bf16x8 hi = *(const bf16x8*)(p + 16);
  v16bf r;
#pragma unroll
  for (int t = 0; t < 8; ++t) { r[t] = lo[t]; r[t + 8] = hi[t]; }
  return r;
}
// LDS variants: slab layout is 128 rows x 32 bf16, padded row stride ROWB bytes.
constexpr int ROWB = 80;                 // 64B data + 16B TDM pad -> conflict-free, 16B aligned
constexpr int SLAB = 128 * ROWB;         // 10240 B per staged slab

__device__ __forceinline__ v16bf lds_a_frag(const char* base, int row, int lane) {
  const char* p = base + (size_t)(row + (lane & 15)) * ROWB + (((lane >> 4) << 3) * 2);
  bf16x8 lo = *(const bf16x8*)p;
  bf16x8 hi = *(const bf16x8*)(p + 32);
  v16bf r;
#pragma unroll
  for (int t = 0; t < 8; ++t) { r[t] = lo[t]; r[t + 8] = hi[t]; }
  return r;
}
__device__ __forceinline__ v16bf lds_b_frag(const char* base, int row, int lane) {
  const char* p = base + (size_t)(row + (lane & 15)) * ROWB + (((lane >> 4) << 3) * 2);
  bf16x8 lo = *(const bf16x8*)p;
  bf16x8 hi = *(const bf16x8*)(p + 32);
  v16bf r;
#pragma unroll
  for (int t = 0; t < 8; ++t) { r[t] = lo[t]; r[t + 8] = hi[t]; }
  return r;
}
#define WMMA_BF16(a, b, c) \
  __builtin_amdgcn_wmma_f32_16x16x32_bf16(false, (a), false, (b), (short)0, (c), false, false)

// ---------------- Tensor Data Mover: 2-D tile (32 bf16 x 128 rows) -> LDS ----------------
// D# per CDNA5 ISA ch.8: group0 = {count/lds_addr/global_addr/type}, group1 = dims/strides.
// pad_enable inserts 4 DWORDs (16B) after every 16 DWORDs (64B = one tile row) -> ROWB=80.
__device__ __forceinline__ void tdm_load_slab(unsigned lds_addr, const bf16_t* gptr) {
  unsigned long long ga = (unsigned long long)(uintptr_t)gptr;
  u32x4 g0;
  g0[0] = 1u;                                             // count=1 (valid), no gather
  g0[1] = lds_addr;                                       // LDS dest (bytes)
  g0[2] = (unsigned)ga;                                   // global addr [31:0]
  g0[3] = (unsigned)((ga >> 32) & 0x01FFFFFFu) | (2u << 30); // addr[56:32] | type=2
  i32x8 g1;
  g1[0] = (1 << 16) | (1 << 20) | (3 << 22) | (3 << 25);  // data_size=2B, pad_en, ivl=16dw, amt=4dw
  g1[1] = (int)((NN & 0xFFFF) << 16);                     // tensor_dim0 lo16 @ [31:16]
  g1[2] = (int)(((NN >> 16) & 0xFFFF) | ((NN & 0xFFFF) << 16)); // tdim0 hi | tdim1 lo
  g1[3] = (int)(((NN >> 16) & 0xFFFF) | (32 << 16));      // tdim1 hi | tile_dim0=32
  g1[4] = 128;                                            // tile_dim1=128, tile_dim2=0
  g1[5] = NN;                                             // tensor_dim0_stride lo32 (elements)
  g1[6] = 0;                                              // stride hi | tdim1_stride lo
  g1[7] = 0;
  i32x4 z = {0, 0, 0, 0};
#if __clang_major__ >= 23
  i32x8 z8 = {0, 0, 0, 0, 0, 0, 0, 0};
  __builtin_amdgcn_tensor_load_to_lds(g0, g1, z, z, z8, 0);
#else
  __builtin_amdgcn_tensor_load_to_lds(g0, g1, z, z, 0);
#endif
}

__device__ __forceinline__ float block_reduce_256(float x, float* red) {
  int t = threadIdx.x;
  red[t] = x; __syncthreads();
  for (int off = 128; off > 0; off >>= 1) {
    if (t < off) red[t] += red[t + off];
    __syncthreads();
  }
  float r = red[0]; __syncthreads();
  return r;
}

// ---------------- elementwise / prep kernels ----------------
__global__ void prep_kernel(const float* __restrict__ X, bf16_t* __restrict__ Xb,
                            float* __restrict__ nsq) {
  __shared__ float red[256];
  int row = blockIdx.x;
  const float* xr = X + (size_t)row * DD;
  bf16_t* br = Xb + (size_t)row * DD;
  float s = 0.f;
  for (int c = threadIdx.x; c < DD; c += 256) {
    float v = xr[c];
    br[c] = (bf16_t)v;
    s += v * v;
  }
  float tot = block_reduce_256(s, red);
  if (threadIdx.x == 0) nsq[row] = tot;
}

__global__ void set_kernel(float* __restrict__ p, int n, float val) {
  int i = blockIdx.x * 256 + threadIdx.x;
  if (i < n) p[i] = val;
}

__global__ void mul_kernel(const float* __restrict__ a, const float* __restrict__ b,
                           float* __restrict__ o, int n) {
  int i = blockIdx.x * 256 + threadIdx.x;
  if (i < n) o[i] = a[i] * b[i];
}

// ---------------- cost-matrix GEMMs (K=256: one wave per 16x16 tile) ----------------
__global__ void cost_cross_kernel(const bf16_t* __restrict__ Ab, const bf16_t* __restrict__ Bb,
                                  const float* __restrict__ n0sq, const float* __restrict__ n1sq,
                                  float* __restrict__ W, float* __restrict__ Km) {
  int wave = blockIdx.x * 8 + (threadIdx.x >> 5);
  int lane = threadIdx.x & 31;
  int i0 = (wave >> 8) << 4;
  int j0 = (wave & 255) << 4;
  v8f acc = {};
#pragma unroll
  for (int k0 = 0; k0 < DD; k0 += 32) {
    v16bf a = load_a_frag(Ab, DD, i0, k0, lane);
    v16bf b = load_bT_frag(Bb, DD, j0, k0, lane);
    acc = WMMA_BF16(a, b, acc);
  }
  int col = j0 + (lane & 15);
  int rb  = i0 + ((lane >> 4) << 3);
  float nc = n1sq[col];
#pragma unroll
  for (int r = 0; r < 8; ++r) {
    int row = rb + r;
    float c = fmaxf(n0sq[row] + nc - 2.0f * acc[r], 0.0f);
    size_t idx = (size_t)row * NN + col;
    W[idx]  = c;
    Km[idx] = __expf(-c * REG_INV);
  }
}

__global__ void cost_self_kernel(const bf16_t* __restrict__ Xb, const float* __restrict__ nsq,
                                 bf16_t* __restrict__ Db, float* __restrict__ rsq) {
  int wave = blockIdx.x * 8 + (threadIdx.x >> 5);
  int lane = threadIdx.x & 31;
  int i0 = (wave >> 8) << 4;
  int j0 = (wave & 255) << 4;
  v8f acc = {};
#pragma unroll
  for (int k0 = 0; k0 < DD; k0 += 32) {
    v16bf a = load_a_frag(Xb, DD, i0, k0, lane);
    v16bf b = load_bT_frag(Xb, DD, j0, k0, lane);
    acc = WMMA_BF16(a, b, acc);
  }
  int col = j0 + (lane & 15);
  int rb  = i0 + ((lane >> 4) << 3);
  float nc = nsq[col];
  float part = 0.f;
#pragma unroll
  for (int r = 0; r < 8; ++r) {
    int row = rb + r;
    float c = fmaxf(nsq[row] + nc - 2.0f * acc[r], 0.0f);
    Db[(size_t)row * NN + col] = (bf16_t)c;
    part += c * c;                       // colSum(D^2) == rowSum(D^2) by symmetry
  }
  atomicAdd(&rsq[col], part);
}

// ---------------- Sinkhorn (K lives in the 192MB L2: matvec-bound) ----------------
__global__ void row_update_u(const float* __restrict__ Km, const float* __restrict__ v,
                             float* __restrict__ u) {
  __shared__ float red[256];
  int row = blockIdx.x;
  const float* kr = Km + (size_t)row * NN;
  float s = 0.f;
  for (int j = threadIdx.x; j < NN; j += 256) s += kr[j] * v[j];
  float tot = block_reduce_256(s, red);
  if (threadIdx.x == 0) u[row] = MU / (tot + EPS_S);
}

__global__ void col_partial(const float* __restrict__ Km, const float* __restrict__ w,
                            float* __restrict__ tmp) {
  int col = blockIdx.x * 256 + threadIdx.x;
  int r0 = blockIdx.y * 128;
  float s = 0.f;
  for (int r = r0; r < r0 + 128; ++r) s += Km[(size_t)r * NN + col] * w[r];
  atomicAdd(&tmp[col], s);
}

__global__ void update_v_kernel(const float* __restrict__ tmp, float* __restrict__ v) {
  int j = blockIdx.x * 256 + threadIdx.x;
  v[j] = MU / (tmp[j] + EPS_S);
}

// ---------------- pi + O(N^2) reductions ----------------
__global__ void pi_build_kernel(const float* __restrict__ Km, const float* __restrict__ W,
                                const float* __restrict__ u, const float* __restrict__ v,
                                bf16_t* __restrict__ pib, float* __restrict__ rsum,
                                float* __restrict__ S) {
  __shared__ float red[256];
  int row = blockIdx.x;
  float ur = u[row];
  const float* kr = Km + (size_t)row * NN;
  const float* wr = W  + (size_t)row * NN;
  bf16_t* pr = pib + (size_t)row * NN;
  float wd = 0.f, rs = 0.f;
  for (int j = threadIdx.x; j < NN; j += 256) {
    float p = ur * kr[j] * v[j];
    pr[j] = (bf16_t)p;
    wd += p * wr[j];
    rs += p;
  }
  float wtot = block_reduce_256(wd, red);
  float rtot = block_reduce_256(rs, red);
  if (threadIdx.x == 0) { rsum[row] = rtot; atomicAdd(&S[0], wtot); }
}

__global__ void term1_kernel(const float* __restrict__ Km, const float* __restrict__ v,
                             const float* __restrict__ csum, const float* __restrict__ u,
                             const float* __restrict__ r1, float* __restrict__ S) {
  __shared__ float red[256];
  int row = blockIdx.x;
  const float* kr = Km + (size_t)row * NN;
  float s = 0.f;
  for (int j = threadIdx.x; j < NN; j += 256) s += kr[j] * (v[j] * csum[j]);
  float tot = block_reduce_256(s, red);
  if (threadIdx.x == 0) atomicAdd(&S[1], r1[row] * u[row] * tot);
}

__global__ void term2_final_kernel(const float* __restrict__ tmp, const float* __restrict__ v,
                                   const float* __restrict__ r2, float* __restrict__ S) {
  __shared__ float red[256];
  int j = blockIdx.x * 256 + threadIdx.x;
  float val = r2[j] * v[j] * tmp[j];
  float tot = block_reduce_256(val, red);
  if (threadIdx.x == 0) atomicAdd(&S[2], tot);
}

// ---------------- transpose pi (so big-GEMM B slabs are row-contiguous) ----------------
__global__ void transpose_bf16_kernel(const bf16_t* __restrict__ in, bf16_t* __restrict__ out) {
  __shared__ bf16_t tile[32][33];
  int bx = blockIdx.x << 5, by = blockIdx.y << 5;
  int tx = threadIdx.x & 31, ty = threadIdx.x >> 5;
  for (int r = ty; r < 32; r += 8)
    tile[r][tx] = in[(size_t)(by + r) * NN + bx + tx];
  __syncthreads();
  for (int r = ty; r < 32; r += 8)
    out[(size_t)(bx + r) * NN + by + tx] = tile[tx][r];
}

// ---------------- big 4096^3 GEMMs: TDM-staged, double-buffered, 128x128/block ----------------
// 8 waves in a 2x4 grid; wave tile 64x32 = 4x2 wmma accumulators.
// STORE_C=1: write C (fp32). STORE_C=0: fuse S += sum(Aref .* C) instead.
template <int STORE_C>
__global__ void __launch_bounds__(256)
gemm_tdm_kernel(const bf16_t* __restrict__ Aglob, const bf16_t* __restrict__ Yglob,
                float* __restrict__ Cout, const float* __restrict__ Aref,
                float* __restrict__ S) {
  __shared__ char sm[4 * SLAB];          // A0 | A1 | B0 | B1  (40 KB)
  __shared__ float red[256];
  int wid  = threadIdx.x >> 5;
  int lane = threadIdx.x & 31;
  int wm = wid >> 2;                     // 0..1  (64-row band)
  int wn = wid & 3;                      // 0..3  (32-col band)
  int i0 = (int)(blockIdx.x >> 5) * 128;
  int j0 = (int)(blockIdx.x & 31) * 128;

  unsigned smA0 = (unsigned)(uintptr_t)&sm[0];
  unsigned smA1 = (unsigned)(uintptr_t)&sm[SLAB];
  unsigned smB0 = (unsigned)(uintptr_t)&sm[2 * SLAB];
  unsigned smB1 = (unsigned)(uintptr_t)&sm[3 * SLAB];

  v8f acc[4][2] = {};

  if (wid == 0) {                        // prologue: stage k-step 0
    tdm_load_slab(smA0, Aglob + (size_t)i0 * NN);
    tdm_load_slab(smB0, Yglob + (size_t)j0 * NN);
  }
  const int NK = NN / 32;
  for (int kt = 0; kt < NK; ++kt) {
    int cur = kt & 1;
    if (wid == 0) {
      if (kt + 1 < NK) {                 // stage k-step kt+1 into the other buffer
        tdm_load_slab(cur ? smA0 : smA1, Aglob + (size_t)i0 * NN + (size_t)(kt + 1) * 32);
        tdm_load_slab(cur ? smB0 : smB1, Yglob + (size_t)j0 * NN + (size_t)(kt + 1) * 32);
        __builtin_amdgcn_s_wait_tensorcnt(2);   // in-order: pair kt complete
      } else {
        __builtin_amdgcn_s_wait_tensorcnt(0);
      }
    }
    __syncthreads();                     // slab kt visible to all waves

    const char* Ab = &sm[cur ? SLAB : 0];
    const char* Bb = &sm[(cur ? 3 : 2) * SLAB];
    v16bf afrag[4], bfrag[2];
#pragma unroll
    for (int m = 0; m < 4; ++m) afrag[m] = lds_a_frag(Ab, wm * 64 + m * 16, lane);
#pragma unroll
    for (int n = 0; n < 2; ++n) bfrag[n] = lds_b_frag(Bb, wn * 32 + n * 16, lane);
#pragma unroll
    for (int m = 0; m < 4; ++m)
#pragma unroll
      for (int n = 0; n < 2; ++n)
        acc[m][n] = WMMA_BF16(afrag[m], bfrag[n], acc[m][n]);

    __syncthreads();                     // all waves done with slab kt before reuse
  }

  float t3 = 0.f;
#pragma unroll
  for (int m = 0; m < 4; ++m)
#pragma unroll
    for (int n = 0; n < 2; ++n) {
      int cc = j0 + wn * 32 + n * 16 + (lane & 15);
      int rb = i0 + wm * 64 + m * 16 + ((lane >> 4) << 3);
#pragma unroll
      for (int r = 0; r < 8; ++r) {
        if (STORE_C) Cout[(size_t)(rb + r) * NN + cc] = acc[m][n][r];
        else         t3 += Aref[(size_t)(rb + r) * NN + cc] * acc[m][n][r];
      }
    }
  if (!STORE_C) {
    float tot = block_reduce_256(t3, red);
    if (threadIdx.x == 0) atomicAdd(&S[3], tot);
  }
}

__global__ void finalize_kernel(const float* __restrict__ S, float* __restrict__ out) {
  float wd = S[0], t1 = S[1], t2 = S[2], t3 = S[3];
  float gw = (t1 + t2 - 2.0f * t3) / ((float)NN * (float)NN);
  out[0] = 0.5f * wd + 0.5f * gw;
}

// ---------------- launch ----------------
extern "C" void kernel_launch(void* const* d_in, const int* in_sizes, int n_in,
                              void* d_out, int out_size, void* d_ws, size_t ws_size,
                              hipStream_t stream) {
  const float* t0 = (const float*)d_in[0];
  const float* t1 = (const float*)d_in[1];
  float* out = (float*)d_out;
  char* ws = (char*)d_ws;

  const size_t MB = (size_t)1 << 20;
  bf16_t* t0b = (bf16_t*)(ws + 0 * MB);
  bf16_t* t1b = (bf16_t*)(ws + 2 * MB);
  float*  vec = (float*)(ws + 4 * MB);
  float* n0sq = vec + 0 * NN;
  float* n1sq = vec + 1 * NN;
  float* u    = vec + 2 * NN;
  float* v    = vec + 3 * NN;
  float* tmp  = vec + 4 * NN;
  float* r1   = vec + 5 * NN;
  float* r2   = vec + 6 * NN;
  float* csum = vec + 7 * NN;
  float* rsum = vec + 8 * NN;
  float* S    = vec + 9 * NN;               // accumulators
  float*  W    = (float*)(ws + 8 * MB);     // 64 MB fp32
  float*  Km   = (float*)(ws + 72 * MB);    // 64 MB fp32
  bf16_t* D1b  = (bf16_t*)(ws + 136 * MB);  // 32 MB bf16
  bf16_t* D2b  = (bf16_t*)(ws + 168 * MB);  // 32 MB bf16
  bf16_t* pib  = (bf16_t*)(ws + 200 * MB);  // 32 MB bf16
  bf16_t* piT  = (bf16_t*)(ws + 232 * MB);  // 32 MB bf16
  float*  Abuf = (float*)(ws + 264 * MB);   // 64 MB fp32 (ends 328 MB)

  dim3 blk(256);
  dim3 gtile(8192);                          // 256x256 tiles of 16x16 (cost GEMMs)
  dim3 gbig(1024);                           // 32x32 block tiles of 128x128 (big GEMMs)

  prep_kernel<<<NN, blk, 0, stream>>>(t0, t0b, n0sq);
  prep_kernel<<<NN, blk, 0, stream>>>(t1, t1b, n1sq);
  set_kernel<<<(4 * NN + 64 + 255) / 256, blk, 0, stream>>>(r1, 4 * NN + 64, 0.0f);
  set_kernel<<<NN / 256, blk, 0, stream>>>(v, NN, 1.0f);

  cost_cross_kernel<<<gtile, blk, 0, stream>>>(t0b, t1b, n0sq, n1sq, W, Km);
  cost_self_kernel<<<gtile, blk, 0, stream>>>(t0b, n0sq, D1b, r1);
  cost_self_kernel<<<gtile, blk, 0, stream>>>(t1b, n1sq, D2b, r2);

  for (int it = 0; it < ITERS; ++it) {
    row_update_u<<<NN, blk, 0, stream>>>(Km, v, u);
    set_kernel<<<NN / 256, blk, 0, stream>>>(tmp, NN, 0.0f);
    col_partial<<<dim3(16, 32), blk, 0, stream>>>(Km, u, tmp);
    update_v_kernel<<<NN / 256, blk, 0, stream>>>(tmp, v);
  }
  // post-loop: tmp == K^T u (final u), v == nu/(tmp+eps)

  pi_build_kernel<<<NN, blk, 0, stream>>>(Km, W, u, v, pib, rsum, S);
  mul_kernel<<<NN / 256, blk, 0, stream>>>(v, tmp, csum, NN);     // csum = colsum(pi)
  term1_kernel<<<NN, blk, 0, stream>>>(Km, v, csum, u, r1, S);
  mul_kernel<<<NN / 256, blk, 0, stream>>>(u, rsum, n0sq, NN);    // wrow = u .* rowsum(pi)
  set_kernel<<<NN / 256, blk, 0, stream>>>(tmp, NN, 0.0f);
  col_partial<<<dim3(16, 32), blk, 0, stream>>>(Km, n0sq, tmp);   // tmp = K^T wrow
  term2_final_kernel<<<NN / 256, blk, 0, stream>>>(tmp, v, r2, S);

  transpose_bf16_kernel<<<dim3(128, 128), blk, 0, stream>>>(pib, piT);
  // Abuf = D1 @ pi   (B slab rows from piT)
  gemm_tdm_kernel<1><<<gbig, blk, 0, stream>>>(D1b, piT, Abuf, nullptr, nullptr);
  // S[3] += sum(Abuf .* (pi @ D2))   (D2 symmetric -> B slab rows from D2b)
  gemm_tdm_kernel<0><<<gbig, blk, 0, stream>>>(pib, D2b, nullptr, Abuf, S);

  finalize_kernel<<<1, 1, 0, stream>>>(S, out);
}
